// HypAgg_7662221656591
// MI455X (gfx1250) — matrix operand order repaired and checked by reading, not compiled, not run on hardware
//
#include <hip/hip_runtime.h>
#include <hip/hip_bf16.h>
#include <math.h>

// HypAgg on the Poincare ball, fused for MI455X (gfx1250, wave32, WMMA).
// B=4, N=384, D=64, c=1. Edge-GEMM runs on v_wmma_f32_16x16x32_bf16.
// Hot loop uses raw TRANS ops: v_exp_f32 / v_log_f32 / v_rcp_f32.

typedef __attribute__((ext_vector_type(16))) __bf16 v16bf;
typedef __attribute__((ext_vector_type(8)))  float  v8f;

#define CB 4
#define CN 384
#define CD 64
#define TIL 16
#define NJT (CN / TIL)          // 24 j-tiles
#define JSPLIT 4                // j-range split for parallelism
#define JT_PER (NJT / JSPLIT)   // 6 j-tiles per block
#define MINN 1e-15f

__device__ __forceinline__ float rcp_f(float v)     { return __builtin_amdgcn_rcpf(v); }
__device__ __forceinline__ float sigmoid_f(float v) { return rcp_f(1.0f + __expf(-v)); }
__device__ __forceinline__ float silu_f(float v)    { return v * sigmoid_f(v); }
// artanh(t) = 0.5*ln((1+t)/(1-t)), t in [0, 1-1e-7]
__device__ __forceinline__ float artanh_f(float t)  { return 0.5f * __logf((1.0f + t) * rcp_f(1.0f - t)); }

// ---------------------------------------------------------------------------
// Kernel 0: zero the support accumulator in d_ws (harness poisons it).
// ---------------------------------------------------------------------------
__global__ void hyp_zero_kernel(float* __restrict__ p, int n) {
    int i = blockIdx.x * blockDim.x + threadIdx.x;
    if (i < n) p[i] = 0.0f;
}

// ---------------------------------------------------------------------------
// Kernel 1: fused edge kernel.
//   For each (b, i-tile, j-chunk): compute u = logmap_{x_i}(x_j) (clipped),
//   h = silu(u @ W1a + b1) via bf16 WMMA, att = sigmoid(h @ w2 + b2),
//   support[i] += sum_j (att*mask) * u.   (u_self == 0 exactly in fp.)
// ---------------------------------------------------------------------------
__global__ __launch_bounds__(256) void hyp_edge_kernel(
    const float* __restrict__ x, const int* __restrict__ mask,
    const float* __restrict__ att_w1, const float* __restrict__ att_b1,
    const float* __restrict__ att_w2, const float* __restrict__ att_b2,
    float* __restrict__ support)
{
    __shared__ float  xi[TIL][CD];
    __shared__ float  xj[TIL][CD];
    __shared__ float  x2i[TIL];
    __shared__ float  x2j[TIL];
    __shared__ __bf16 w1a[CD][CD];          // att_w1 rows 0..63 (bf16)
    __shared__ float  b1s[CD];
    __shared__ float  w2s[CD];
    __shared__ __bf16 uT[TIL][TIL][CD];     // [i_local][j_local][d], clipped u
    __shared__ float  wtile[TIL][TIL];      // att * mask per edge
    __shared__ float  supAcc[TIL][CD];

    const int tid = threadIdx.x;
    const int wv  = tid >> 5;       // wave id 0..7
    const int ln  = tid & 31;
    const int lo  = ln & 15;        // j_local / N-lane
    const int hi  = ln >> 4;        // d-half / lane-half

    int bx = blockIdx.x;
    const int jchunk = bx % JSPLIT; bx /= JSPLIT;
    const int it     = bx % NJT;    bx /= NJT;
    const int b      = bx;
    const int i0     = it * TIL;

    const float b2 = att_b2[0];

    for (int k = tid; k < TIL * CD; k += 256) {
        int r = k >> 6, d = k & 63;
        xi[r][d]     = x[((size_t)b * CN + i0 + r) * CD + d];
        supAcc[r][d] = 0.0f;
    }
    for (int k = tid; k < CD * CD; k += 256) {
        int r = k >> 6, n = k & 63;
        w1a[r][n] = (__bf16)att_w1[r * CD + n];   // rows 0..63 only (u_self==0)
    }
    if (tid < CD) { b1s[tid] = att_b1[tid]; w2s[tid] = att_w2[tid]; }
    __syncthreads();
    if (tid < TIL) {
        float s = 0.0f;
        for (int d = 0; d < CD; ++d) s += xi[tid][d] * xi[tid][d];
        x2i[tid] = s;
    }
    __syncthreads();

    // Preload B-fragments of W1a in registers (contiguous k order, same as A:
    // lane: N = nc*16+lo, element e -> K = kk*32 + hi*16 + e).
    v16bf bfr[4][2];
    #pragma unroll
    for (int nc = 0; nc < 4; ++nc)
        #pragma unroll
        for (int kk = 0; kk < 2; ++kk)
            #pragma unroll
            for (int e = 0; e < 16; ++e)
                bfr[nc][kk][e] = w1a[kk * 32 + hi * 16 + e][nc * 16 + lo];

    for (int jt = 0; jt < JT_PER; ++jt) {
        const int j0 = (jchunk * JT_PER + jt) * TIL;
        __syncthreads();   // previous phase-2 done before overwriting xj/uT
        for (int k = tid; k < TIL * CD; k += 256) {
            int r = k >> 6, d = k & 63;
            xj[r][d] = x[((size_t)b * CN + j0 + r) * CD + d];
        }
        __syncthreads();
        if (tid < TIL) {
            float s = 0.0f;
            for (int d = 0; d < CD; ++d) s += xj[tid][d] * xj[tid][d];
            x2j[tid] = s;
        }
        __syncthreads();

        // ---- phase 1: each wave owns 2 i-rows; 2 lanes per edge split D ----
        for (int sub = 0; sub < 2; ++sub) {
            const int il  = wv * 2 + sub;
            const float X2i = x2i[il];
            const float X2j = x2j[lo];

            float dotp = 0.0f;
            #pragma unroll
            for (int dd = 0; dd < 32; ++dd) {
                int d = hi * 32 + dd;
                dotp += xi[il][d] * xj[lo][d];
            }
            const float dot = dotp + __shfl_xor(dotp, 16, 32);

            // mobius_add(-x_i, x_j), c=1:
            const float ca  = 1.0f - 2.0f * dot + X2j;    // coeff on x_i (negated)
            const float cb  = 1.0f - X2i;                 // coeff on x_j
            const float den = 1.0f - 2.0f * dot + X2i * X2j;
            const float invden = rcp_f(fmaxf(den, MINN));

            float s[32];
            float sn2p = 0.0f;
            #pragma unroll
            for (int dd = 0; dd < 32; ++dd) {
                int d = hi * 32 + dd;
                float sv = (cb * xj[lo][d] - ca * xi[il][d]) * invden;
                s[dd] = sv;
                sn2p += sv * sv;
            }
            const float sn2 = sn2p + __shfl_xor(sn2p, 16, 32);
            const float sn  = fmaxf(sqrtf(sn2), MINN);
            const float two_over_lam = fmaxf(1.0f - X2i, MINN);  // 2/lam
            const float t     = fminf(sn, 1.0f - 1e-7f);
            const float art   = artanh_f(t);
            const float scale = two_over_lam * art * rcp_f(sn);

            #pragma unroll
            for (int dd = 0; dd < 32; ++dd) {
                float u = scale * s[dd];
                u = fminf(fmaxf(u, -1000.0f), 1000.0f);
                uT[il][lo][hi * 32 + dd] = (__bf16)u;
            }
            __syncthreads();   // uniform: make uT visible for A-frag loads

            // A-fragment (contiguous k order, same as B): lane row M = lo (= j_local),
            // element e -> K = hi*16 + e  [+32 for a1].
            v16bf a0, a1;
            #pragma unroll
            for (int e = 0; e < 16; ++e) {
                int k = hi * 16 + e;
                a0[e] = uT[il][lo][k];
                a1[e] = uT[il][lo][32 + k];
            }

            v8f z = {0.f,0.f,0.f,0.f,0.f,0.f,0.f,0.f};
            v8f acc[4] = {z, z, z, z};
            #pragma unroll
            for (int nc = 0; nc < 4; ++nc) {
                acc[nc] = __builtin_amdgcn_wmma_f32_16x16x32_bf16(
                    false, a0, false, bfr[nc][0], (short)0, acc[nc], false, false);
                acc[nc] = __builtin_amdgcn_wmma_f32_16x16x32_bf16(
                    false, a1, false, bfr[nc][1], (short)0, acc[nc], false, false);
            }

            // D-frag: lane holds (M = r + 8*hi, N = nc*16 + lo).
            // att logit per edge row: sum_n silu(h[m][n] + b1[n]) * w2[n].
            float attp[8];
            #pragma unroll
            for (int r = 0; r < 8; ++r) attp[r] = 0.0f;
            #pragma unroll
            for (int nc = 0; nc < 4; ++nc) {
                const int   n   = nc * 16 + lo;
                const float bb  = b1s[n];
                const float w2v = w2s[n];
                #pragma unroll
                for (int r = 0; r < 8; ++r) {
                    float h = acc[nc][r] + bb;
                    attp[r] += silu_f(h) * w2v;
                }
            }
            #pragma unroll
            for (int m = 1; m < 16; m <<= 1)
                #pragma unroll
                for (int r = 0; r < 8; ++r)
                    attp[r] += __shfl_xor(attp[r], m, 32);

            if (lo == 0) {
                #pragma unroll
                for (int r = 0; r < 8; ++r) {
                    int j = r + 8 * hi;
                    float a  = sigmoid_f(attp[r] + b2);
                    float mk = (float)mask[((size_t)b * CN + (i0 + il)) * CN + (j0 + j)];
                    wtile[il][j] = a * mk;
                }
            }
        } // sub
        __syncthreads();

        // ---- phase 2: support[i][d] += sum_j w[i][j] * u[i][j][d] ----
        {
            const int il    = tid >> 4;
            const int dbase = (tid & 15) * 4;
            #pragma unroll
            for (int dd = 0; dd < 4; ++dd) {
                int d = dbase + dd;
                float a = 0.0f;
                #pragma unroll
                for (int j = 0; j < TIL; ++j)
                    a += wtile[il][j] * (float)uT[il][j][d];
                supAcc[il][d] += a;
            }
        }
    } // jt

    __syncthreads();
    for (int k = tid; k < TIL * CD; k += 256) {
        int r = k >> 6, d = k & 63;
        atomicAdd(&support[((size_t)b * CN + i0 + r) * CD + d], supAcc[r][d]);
    }
}

// ---------------------------------------------------------------------------
// Kernel 2: node MLP + expmap + proj.  feat = concat(u_self==0, support), so
// out_t = silu(support @ mlp_w1[64:] + b1) @ mlp_w2 + b2;  then
// out = proj(expmap(out_t, x)).  (Cold kernel: keep precise math.)
// ---------------------------------------------------------------------------
__global__ __launch_bounds__(256) void hyp_node_kernel(
    const float* __restrict__ x, const float* __restrict__ support,
    const float* __restrict__ mlp_w1, const float* __restrict__ mlp_b1,
    const float* __restrict__ mlp_w2, const float* __restrict__ mlp_b2,
    float* __restrict__ out)
{
    __shared__ float sup[TIL][CD];
    __shared__ float hmid[TIL][CD];
    __shared__ float outt[TIL][CD];
    __shared__ float xi[TIL][CD];
    __shared__ float res[TIL][CD];
    __shared__ float sc[TIL][4];
    __shared__ float pf[TIL];

    const int tid = threadIdx.x;
    int bx = blockIdx.x;
    const int it = bx % NJT;
    const int b  = bx / NJT;
    const int i0 = it * TIL;

    for (int k = tid; k < TIL * CD; k += 256) {
        int r = k >> 6, d = k & 63;
        sup[r][d] = support[((size_t)b * CN + i0 + r) * CD + d];
        xi[r][d]  = x[((size_t)b * CN + i0 + r) * CD + d];
    }
    __syncthreads();
    {
        const int r   = tid >> 4;
        const int cb4 = (tid & 15) * 4;
        for (int k = 0; k < 4; ++k) {
            int hc = cb4 + k;
            float a = mlp_b1[hc];
            for (int d = 0; d < CD; ++d)
                a += sup[r][d] * mlp_w1[(CD + d) * CD + hc];
            hmid[r][hc] = a / (1.0f + expf(-a));
        }
    }
    __syncthreads();
    {
        const int r   = tid >> 4;
        const int cb4 = (tid & 15) * 4;
        for (int k = 0; k < 4; ++k) {
            int o = cb4 + k;
            float a = mlp_b2[o];
            for (int h = 0; h < CD; ++h)
                a += hmid[r][h] * mlp_w2[h * CD + o];
            outt[r][o] = a;   // + u_self == 0
        }
    }
    __syncthreads();
    if (tid < TIL) {
        float un2 = 0.0f, x2 = 0.0f, dxy = 0.0f;
        for (int d = 0; d < CD; ++d) {
            float u = outt[tid][d], xv = xi[tid][d];
            un2 += u * u; x2 += xv * xv; dxy += u * xv;
        }
        float un  = fmaxf(sqrtf(un2), MINN);
        float lam = 2.0f / fmaxf(1.0f - x2, MINN);
        float th  = tanhf(0.5f * lam * un);
        float ss  = th / un;                 // second = ss * out_t
        float y2  = ss * ss * un2;
        float xy  = ss * dxy;
        float den = 1.0f + 2.0f * xy + x2 * y2;
        sc[tid][0] = ss;
        sc[tid][1] = 1.0f + 2.0f * xy + y2;  // coeff on x
        sc[tid][2] = 1.0f - x2;              // coeff on second
        sc[tid][3] = 1.0f / fmaxf(den, MINN);
    }
    __syncthreads();
    for (int k = tid; k < TIL * CD; k += 256) {
        int r = k >> 6, d = k & 63;
        float sec = sc[r][0] * outt[r][d];
        res[r][d] = (sc[r][1] * xi[r][d] + sc[r][2] * sec) * sc[r][3];
    }
    __syncthreads();
    if (tid < TIL) {
        float n2 = 0.0f;
        for (int d = 0; d < CD; ++d) n2 += res[tid][d] * res[tid][d];
        float nrm = fmaxf(sqrtf(n2), MINN);
        const float maxnorm = 1.0f - 4e-3f;
        pf[tid] = (nrm > maxnorm) ? (maxnorm / nrm) : 1.0f;
    }
    __syncthreads();
    for (int k = tid; k < TIL * CD; k += 256) {
        int r = k >> 6, d = k & 63;
        out[((size_t)b * CN + i0 + r) * CD + d] = res[r][d] * pf[r];
    }
}

// ---------------------------------------------------------------------------
extern "C" void kernel_launch(void* const* d_in, const int* in_sizes, int n_in,
                              void* d_out, int out_size, void* d_ws, size_t ws_size,
                              hipStream_t stream) {
    (void)in_sizes; (void)n_in; (void)out_size; (void)ws_size;
    const float* x      = (const float*)d_in[0];
    const int*   mask   = (const int*)  d_in[1];
    const float* att_w1 = (const float*)d_in[2];
    const float* att_b1 = (const float*)d_in[3];
    const float* att_w2 = (const float*)d_in[4];
    const float* att_b2 = (const float*)d_in[5];
    const float* mlp_w1 = (const float*)d_in[6];
    const float* mlp_b1 = (const float*)d_in[7];
    const float* mlp_w2 = (const float*)d_in[8];
    const float* mlp_b2 = (const float*)d_in[9];
    float* out     = (float*)d_out;
    float* support = (float*)d_ws;     // B*N*D f32 scratch

    const int nsup = CB * CN * CD;
    hyp_zero_kernel<<<(nsup + 255) / 256, 256, 0, stream>>>(support, nsup);
    hyp_edge_kernel<<<CB * NJT * JSPLIT, 256, 0, stream>>>(
        x, mask, att_w1, att_b1, att_w2, att_b2, support);
    hyp_node_kernel<<<CB * NJT, 256, 0, stream>>>(
        x, support, mlp_w1, mlp_b1, mlp_w2, mlp_b2, out);
}
